// SMambaNetwork_52656299049424
// MI455X (gfx1250) — matrix operand order repaired and checked by reading, not compiled, not run on hardware
//
#include <hip/hip_runtime.h>
#include <math.h>

// ---------------- problem constants (must match reference) ----------------
#define Bq 16
#define Lq 512
#define Nq 512
#define DM 512
#define DSq 16
#define NLq 2
#define DFF 2048
#define DI 1024      // 2*DM
#define DCONV 4
#define DTR 32       // DM/16
#define Hq 96
#define EPSq 1e-5f
#define Sq Nq        // mamba sequence length = number of variates

typedef __attribute__((ext_vector_type(16))) _Float16 v16h;
typedef __attribute__((ext_vector_type(8)))  float    v8f;

// ---------------- device helpers ----------------
__device__ __forceinline__ float silu_f(float x) {
  return x * (1.0f / (1.0f + __expf(-x)));
}
__device__ __forceinline__ float gelu_f(float x) {
  // jax.nn.gelu approximate=True (tanh)
  float x3 = x * x * x;
  return 0.5f * x * (1.0f + tanhf(0.7978845608028654f * (x + 0.044715f * x3)));
}
__device__ __forceinline__ float softplus_f(float x) {
  return (x > 20.0f) ? x : log1pf(__expf(x));
}

// ---------------- WMMA GEMM: C[M,N] = act(A[M,K] * W[N,K]^T + bias) (+= optional) ----
// A row-major stride lda, W row-major (N rows over K) stride ldw.
// Block tile 64(M) x 64(N), 8 waves (4x2); each wave computes 16x32 via two
// v_wmma_f32_16x16x32_f16 sharing the A fragment. K step 32.
// Double-buffered LDS: next tile's 4x global_load_b128 are issued before the
// current tile's fragment reads + WMMAs, stashed (fp32->f16 packed, one b128
// LDS store each) into the alternate buffer; one barrier per K-step.
// M%64==0 and K%32==0 at every call site; N may be ragged (head N=96) ->
// clamped W loads (uniform flow, EXEC stays all-ones) + guarded stores.
#define ACT_NONE 0
#define ACT_GELU 1
#define ACT_SOFTPLUS 2

#define LDA_ST 40   // halves per LDS row: even (dword pairs) and 80B (16B-aligned rows)

__global__ __launch_bounds__(256)
void gemm_wmma_kernel(const float* __restrict__ A, int lda,
                      const float* __restrict__ W, int ldw,
                      const float* __restrict__ bias,
                      float* __restrict__ C, int ldc,
                      int M, int Nn, int K, int act, int accum)
{
  __shared__ _Float16 As[2][64 * LDA_ST];
  __shared__ _Float16 Ws[2][64 * LDA_ST];

  const int tid  = threadIdx.x;
  const int lane = tid & 31;
  const int wave = tid >> 5;
  const int wm   = wave >> 1;      // 0..3
  const int wn   = wave & 1;       // 0..1

  const int m0 = blockIdx.y * 64;
  const int n0 = blockIdx.x * 64;

  const int khalf = lane >> 4;     // 0..1
  const int l15   = lane & 15;

  const int am  = wm * 16 + l15;        // A row (local) owned by this lane
  const int bn0 = wn * 32 + l15;        // W rows (local) for the two sub-tiles
  const int bn1 = wn * 32 + 16 + l15;

  // cooperative-load geometry: thread -> (row, 8-wide column chunk)
  const int lr = tid >> 2;              // 0..63
  const int lc = (tid & 3) * 8;         // 0,8,16,24
  const int arow = m0 + lr;                                  // M % 64 == 0
  const int wrow = (n0 + lr < Nn) ? (n0 + lr) : (Nn - 1);    // clamp ragged N

  const float* Ap = A + (size_t)arow * lda + lc;
  const float* Wp = W + (size_t)wrow * ldw + lc;

  v8f acc0 = {}, acc1 = {};
  float4 ra0, ra1, rw0, rw1;            // in-flight prefetch registers

  // prefetch + stash tile 0
  ra0 = *(const float4*)(Ap + 0);
  ra1 = *(const float4*)(Ap + 4);
  rw0 = *(const float4*)(Wp + 0);
  rw1 = *(const float4*)(Wp + 4);
  {
    union { _Float16 h[8]; uint4 u; } pa, pw;
    pa.h[0] = (_Float16)ra0.x; pa.h[1] = (_Float16)ra0.y;
    pa.h[2] = (_Float16)ra0.z; pa.h[3] = (_Float16)ra0.w;
    pa.h[4] = (_Float16)ra1.x; pa.h[5] = (_Float16)ra1.y;
    pa.h[6] = (_Float16)ra1.z; pa.h[7] = (_Float16)ra1.w;
    pw.h[0] = (_Float16)rw0.x; pw.h[1] = (_Float16)rw0.y;
    pw.h[2] = (_Float16)rw0.z; pw.h[3] = (_Float16)rw0.w;
    pw.h[4] = (_Float16)rw1.x; pw.h[5] = (_Float16)rw1.y;
    pw.h[6] = (_Float16)rw1.z; pw.h[7] = (_Float16)rw1.w;
    *(uint4*)&As[0][lr * LDA_ST + lc] = pa.u;
    *(uint4*)&Ws[0][lr * LDA_ST + lc] = pw.u;
  }
  __syncthreads();

  const int ntiles = K >> 5;
  for (int i = 0; i < ntiles; ++i) {
    const int buf = i & 1;

    // issue next tile's global loads before consuming this tile (latency overlap)
    if (i + 1 < ntiles) {
      const int k1 = (i + 1) << 5;
      ra0 = *(const float4*)(Ap + k1);
      ra1 = *(const float4*)(Ap + k1 + 4);
      rw0 = *(const float4*)(Wp + k1);
      rw1 = *(const float4*)(Wp + k1 + 4);
    }

    // ---- fragments per documented 16-bit layouts (A 16x32, B 32x16)
    union { v16h v; unsigned u[8]; } Af, Bf0, Bf1;
    const unsigned* Arow = (const unsigned*)&As[buf][am  * LDA_ST];
    const unsigned* Wr0  = (const unsigned*)&Ws[buf][bn0 * LDA_ST];
    const unsigned* Wr1  = (const unsigned*)&Ws[buf][bn1 * LDA_ST];
    #pragma unroll
    for (int j = 0; j < 8; ++j) {
      int kA = 8 * khalf + ((j < 4) ? (2 * j) : (16 + 2 * (j - 4)));
      Af.u[j] = Arow[kA >> 1];
      int kB = 8 * khalf + ((j < 4) ? (2 * j) : (16 + 2 * (j - 4)));
      Bf0.u[j] = Wr0[kB >> 1];
      Bf1.u[j] = Wr1[kB >> 1];
    }
    acc0 = __builtin_amdgcn_wmma_f32_16x16x32_f16(
        false, Af.v, false, Bf0.v, (short)0, acc0, false, false);
    acc1 = __builtin_amdgcn_wmma_f32_16x16x32_f16(
        false, Af.v, false, Bf1.v, (short)0, acc1, false, false);

    // stash prefetched tile into the alternate buffer, then one barrier
    if (i + 1 < ntiles) {
      union { _Float16 h[8]; uint4 u; } pa, pw;
      pa.h[0] = (_Float16)ra0.x; pa.h[1] = (_Float16)ra0.y;
      pa.h[2] = (_Float16)ra0.z; pa.h[3] = (_Float16)ra0.w;
      pa.h[4] = (_Float16)ra1.x; pa.h[5] = (_Float16)ra1.y;
      pa.h[6] = (_Float16)ra1.z; pa.h[7] = (_Float16)ra1.w;
      pw.h[0] = (_Float16)rw0.x; pw.h[1] = (_Float16)rw0.y;
      pw.h[2] = (_Float16)rw0.z; pw.h[3] = (_Float16)rw0.w;
      pw.h[4] = (_Float16)rw1.x; pw.h[5] = (_Float16)rw1.y;
      pw.h[6] = (_Float16)rw1.z; pw.h[7] = (_Float16)rw1.w;
      *(uint4*)&As[buf ^ 1][lr * LDA_ST + lc] = pa.u;
      *(uint4*)&Ws[buf ^ 1][lr * LDA_ST + lc] = pw.u;
      __syncthreads();
    }
  }

  // ---- store: C/D layout — lane col = l15, rows r (+8 for upper lane half)
  const int mbase = m0 + wm * 16 + khalf * 8;
  #pragma unroll
  for (int sub = 0; sub < 2; ++sub) {
    const int n = n0 + wn * 32 + sub * 16 + l15;
    if (n < Nn) {
      float bv = bias ? bias[n] : 0.0f;
      const v8f& acc = sub ? acc1 : acc0;
      #pragma unroll
      for (int r = 0; r < 8; ++r) {
        float v = acc[r] + bv;
        if (act == ACT_GELU)          v = gelu_f(v);
        else if (act == ACT_SOFTPLUS) v = softplus_f(v);
        size_t off = (size_t)(mbase + r) * ldc + n;
        if (accum) C[off] += v; else C[off] = v;
      }
    }
  }
}

// ---------------- RevIN stats: per (b,n) mean & 1/std over L ----------------
__global__ void revin_stats_kernel(const float* __restrict__ x,
                                   float* __restrict__ mu,
                                   float* __restrict__ sd,
                                   float* __restrict__ isd)
{
  int idx = blockIdx.x * 256 + threadIdx.x;       // b*N + n
  if (idx >= Bq * Nq) return;
  int b = idx / Nq, n = idx % Nq;
  const float* p = x + (size_t)b * Lq * Nq + n;
  float s = 0.f, s2 = 0.f;
  for (int l = 0; l < Lq; ++l) { float v = p[(size_t)l * Nq]; s += v; s2 += v * v; }
  float m = s / Lq;
  float var = s2 / Lq - m * m;
  float sdv = sqrtf(var + EPSq);
  mu[idx] = m; sd[idx] = sdv; isd[idx] = 1.0f / sdv;
}

// ---------------- normalized transpose: xt[b,n,l] = (x[b,l,n]-mu)/sd ------------
__global__ void transpose_norm_kernel(const float* __restrict__ x,
                                      const float* __restrict__ mu,
                                      const float* __restrict__ isd,
                                      float* __restrict__ xt)
{
  __shared__ float tile[32][33];
  int b = blockIdx.z;
  int l0 = blockIdx.x * 32, n0 = blockIdx.y * 32;
  int tx = threadIdx.x, ty = threadIdx.y;          // (32,8)
  #pragma unroll
  for (int i = 0; i < 32; i += 8)
    tile[ty + i][tx] = x[(size_t)b * Lq * Nq + (size_t)(l0 + ty + i) * Nq + (n0 + tx)];
  __syncthreads();
  #pragma unroll
  for (int i = 0; i < 32; i += 8) {
    int n = n0 + ty + i, l = l0 + tx;
    int bn = b * Nq + n;
    xt[(size_t)b * Nq * Lq + (size_t)n * Lq + l] = (tile[tx][ty + i] - mu[bn]) * isd[bn];
  }
}

// ---------------- LayerNorm over last dim D=512, one block per row -------------
__global__ __launch_bounds__(256)
void layernorm_kernel(const float* __restrict__ x, const float* __restrict__ w,
                      const float* __restrict__ b, float* __restrict__ y)
{
  int row = blockIdx.x;
  const float* xr = x + (size_t)row * DM;
  float s = 0.f, s2 = 0.f;
  for (int i = threadIdx.x; i < DM; i += 256) { float v = xr[i]; s += v; s2 += v * v; }
  #pragma unroll
  for (int off = 16; off > 0; off >>= 1) {
    s  += __shfl_down(s,  off, 32);
    s2 += __shfl_down(s2, off, 32);
  }
  __shared__ float red0[8], red1[8];
  int wave = threadIdx.x >> 5, lane = threadIdx.x & 31;
  if (lane == 0) { red0[wave] = s; red1[wave] = s2; }
  __syncthreads();
  if (wave == 0) {
    s  = (lane < 8) ? red0[lane] : 0.f;
    s2 = (lane < 8) ? red1[lane] : 0.f;
    #pragma unroll
    for (int off = 4; off > 0; off >>= 1) {
      s  += __shfl_down(s,  off, 32);
      s2 += __shfl_down(s2, off, 32);
    }
    if (lane == 0) { red0[0] = s; red1[0] = s2; }
  }
  __syncthreads();
  float mean = red0[0] / DM;
  float var  = red1[0] / DM - mean * mean;
  float inv  = rsqrtf(var + EPSq);
  for (int i = threadIdx.x; i < DM; i += 256)
    y[(size_t)row * DM + i] = (xr[i] - mean) * inv * w[i] + b[i];
}

// ---------------- causal depthwise conv (k=4) + SiLU -------------------------
// xz: (B*S, 2*DI) fp32, channels [0,DI) are u.  reverse flips tap direction.
__global__ void conv_silu_kernel(const float* __restrict__ xz,
                                 const float* __restrict__ cw,
                                 const float* __restrict__ cb,
                                 float* __restrict__ u, int reverse)
{
  int idx = blockIdx.x * 256 + threadIdx.x;        // (b*S+s)*DI + c
  int c = idx % DI;
  int bs = idx / DI;
  int s = bs % Sq, b = bs / Sq;
  float acc = cb[c];
  #pragma unroll
  for (int j = 0; j < DCONV; ++j) {
    int sp = reverse ? (s + 3 - j) : (s - 3 + j);
    if (sp >= 0 && sp < Sq)
      acc += xz[(size_t)(b * Sq + sp) * (2 * DI) + c] * cw[c * DCONV + j];
  }
  u[idx] = silu_f(acc);
}

// ---------------- selective scan: one thread owns a channel d, 16 states -----
__global__ __launch_bounds__(128)
void scan_kernel(const float* __restrict__ delta, const float* __restrict__ u,
                 const float* __restrict__ xdbl, const float* __restrict__ A_log,
                 float* __restrict__ ys, int reverse)
{
  const int blocksPerB = DI / 128;
  int d = (blockIdx.x % blocksPerB) * 128 + threadIdx.x;
  int b = blockIdx.x / blocksPerB;
  float Ad[DSq], h[DSq];
  #pragma unroll
  for (int i = 0; i < DSq; ++i) { Ad[i] = -__expf(A_log[d * DSq + i]); h[i] = 0.f; }
  for (int t = 0; t < Sq; ++t) {
    int s = reverse ? (Sq - 1 - t) : t;
    size_t row = (size_t)b * Sq + s;
    float dlt = delta[row * DI + d];
    float uv  = u[row * DI + d];
    float du  = dlt * uv;
    const float4* Bm = (const float4*)(xdbl + row * 64 + DTR);
    const float4* Cm = (const float4*)(xdbl + row * 64 + DTR + DSq);
    float y = 0.f;
    #pragma unroll
    for (int q = 0; q < 4; ++q) {
      float4 Bv = Bm[q], Cv = Cm[q];
      h[4*q+0] = __expf(dlt * Ad[4*q+0]) * h[4*q+0] + du * Bv.x; y += h[4*q+0] * Cv.x;
      h[4*q+1] = __expf(dlt * Ad[4*q+1]) * h[4*q+1] + du * Bv.y; y += h[4*q+1] * Cv.y;
      h[4*q+2] = __expf(dlt * Ad[4*q+2]) * h[4*q+2] + du * Bv.z; y += h[4*q+2] * Cv.z;
      h[4*q+3] = __expf(dlt * Ad[4*q+3]) * h[4*q+3] + du * Bv.w; y += h[4*q+3] * Cv.w;
    }
    ys[row * DI + d] = y;
  }
}

// ---------------- gate: ys = (ys + u*D) * silu(z) (in place) -----------------
__global__ void ygate_kernel(float* __restrict__ ys, const float* __restrict__ u,
                             const float* __restrict__ xz, const float* __restrict__ Dp)
{
  int idx = blockIdx.x * 256 + threadIdx.x;
  int c = idx % DI;
  size_t row = (size_t)(idx / DI);
  float z = xz[row * (2 * DI) + DI + c];
  ys[idx] = (ys[idx] + u[idx] * Dp[c]) * silu_f(z);
}

// ---------------- RevIN inverse + transpose: out[b,h,n] -----------------------
__global__ void head_out_kernel(const float* __restrict__ pred,
                                const float* __restrict__ sd,
                                const float* __restrict__ mu,
                                float* __restrict__ out)
{
  int idx = blockIdx.x * 256 + threadIdx.x;        // b*H*N + h*N + n
  if (idx >= Bq * Hq * Nq) return;
  int n = idx % Nq;
  int bh = idx / Nq;
  int h = bh % Hq, b = bh / Hq;
  int bn = b * Nq + n;
  out[idx] = pred[(size_t)bn * Hq + h] * sd[bn] + mu[bn];
}

// ---------------- host-side launch helpers ----------------
static inline void gemm(hipStream_t st, const float* A, int lda, const float* W, int ldw,
                        const float* bias, float* C, int ldc,
                        int M, int Nn, int K, int act, int accum)
{
  dim3 grid((Nn + 63) / 64, M / 64), block(256);
  gemm_wmma_kernel<<<grid, block, 0, st>>>(A, lda, W, ldw, bias, C, ldc, M, Nn, K, act, accum);
}

extern "C" void kernel_launch(void* const* d_in, const int* in_sizes, int n_in,
                              void* d_out, int out_size, void* d_ws, size_t ws_size,
                              hipStream_t stream)
{
  const float* x       = (const float*)d_in[0];
  const float* tok_w   = (const float*)d_in[1];
  const float* tok_b   = (const float*)d_in[2];
  const float* in_w    = (const float*)d_in[3];
  const float* in_b    = (const float*)d_in[4];
  const float* conv_w  = (const float*)d_in[5];
  const float* conv_b  = (const float*)d_in[6];
  const float* xp_w    = (const float*)d_in[7];
  const float* dt_w    = (const float*)d_in[8];
  const float* dt_b    = (const float*)d_in[9];
  const float* A_log   = (const float*)d_in[10];
  const float* Dm      = (const float*)d_in[11];
  const float* out_w   = (const float*)d_in[12];
  const float* out_b   = (const float*)d_in[13];
  const float* mnorm_w = (const float*)d_in[14];
  const float* mnorm_b = (const float*)d_in[15];
  const float* fnorm_w = (const float*)d_in[16];
  const float* fnorm_b = (const float*)d_in[17];
  const float* ffn_w1  = (const float*)d_in[18];
  const float* ffn_b1  = (const float*)d_in[19];
  const float* ffn_w2  = (const float*)d_in[20];
  const float* ffn_b2  = (const float*)d_in[21];
  const float* norm_w  = (const float*)d_in[22];
  const float* norm_b  = (const float*)d_in[23];
  const float* head_w  = (const float*)d_in[24];
  const float* head_b  = (const float*)d_in[25];
  float* out = (float*)d_out;

  const int M = Bq * Sq;                // 8192 rows everywhere

  // workspace bump allocation (floats)
  float* f = (float*)d_ws;
  float* mu  = f;  f += Bq * Nq;
  float* sd  = f;  f += Bq * Nq;
  float* isd = f;  f += Bq * Nq;
  float* tokens = f; f += (size_t)M * DM;
  float* yn     = f; f += (size_t)M * DM;
  float* ubuf   = f; f += (size_t)M * DI;
  float* xdbl   = f; f += (size_t)M * (DTR + 2 * DSq);
  float* delta  = f; f += (size_t)M * DI;
  float* ys     = f; f += (size_t)M * DI;
  float* pred   = f; f += (size_t)M * Hq;
  float* SCR    = f; f += (size_t)M * (2 * DI);    // shared: xt / xz / ffn hidden
  (void)ws_size; (void)n_in; (void)in_sizes; (void)out_size;

  // 1) RevIN stats + normalized transpose into SCR (used as xt: B,N,L)
  revin_stats_kernel<<<(Bq * Nq + 255) / 256, 256, 0, stream>>>(x, mu, sd, isd);
  {
    dim3 g(Lq / 32, Nq / 32, Bq), blk(32, 8);
    transpose_norm_kernel<<<g, blk, 0, stream>>>(x, mu, isd, SCR);
  }
  // 2) variate-token embedding: tokens = xt @ tok_w^T + tok_b
  gemm(stream, SCR, Lq, tok_w, Lq, tok_b, tokens, DM, M, DM, Lq, ACT_NONE, 0);

  for (int l = 0; l < NLq; ++l) {
    // mamba-branch LN (shared by both directions)
    layernorm_kernel<<<M, 256, 0, stream>>>(tokens, mnorm_w + l * DM, mnorm_b + l * DM, yn);

    for (int dir = 0; dir < 2; ++dir) {
      int w2 = l * 2 + dir;
      const float* inw  = in_w  + (size_t)w2 * (2 * DI) * DM;
      const float* inb  = in_b  + (size_t)w2 * (2 * DI);
      const float* cw   = conv_w + (size_t)w2 * DI * DCONV;
      const float* cb   = conv_b + (size_t)w2 * DI;
      const float* xpw  = xp_w  + (size_t)w2 * (DTR + 2 * DSq) * DI;
      const float* dtw  = dt_w  + (size_t)w2 * DI * DTR;
      const float* dtb  = dt_b  + (size_t)w2 * DI;
      const float* Al   = A_log + (size_t)w2 * DI * DSq;
      const float* Dp   = Dm    + (size_t)w2 * DI;
      const float* ow   = out_w + (size_t)w2 * DM * DI;
      const float* ob   = out_b + (size_t)w2 * DM;

      // in_proj: xz (M x 2DI) into SCR
      gemm(stream, yn, DM, inw, DM, inb, SCR, 2 * DI, M, 2 * DI, DM, ACT_NONE, 0);
      // depthwise conv + SiLU -> u
      conv_silu_kernel<<<(M * DI) / 256, 256, 0, stream>>>(SCR, cw, cb, ubuf, dir);
      // x_proj: x_dbl (M x 64), no bias
      gemm(stream, ubuf, DI, xpw, DI, nullptr, xdbl, DTR + 2 * DSq, M, DTR + 2 * DSq, DI, ACT_NONE, 0);
      // dt_proj + softplus -> delta (A = x_dbl[:, :32] via lda)
      gemm(stream, xdbl, DTR + 2 * DSq, dtw, DTR, dtb, delta, DI, M, DI, DTR, ACT_SOFTPLUS, 0);
      // selective scan -> ys
      scan_kernel<<<Bq * (DI / 128), 128, 0, stream>>>(delta, ubuf, xdbl, Al, ys, dir);
      // gate: ys = (ys + u*D) * silu(z)
      ygate_kernel<<<(M * DI) / 256, 256, 0, stream>>>(ys, ubuf, SCR, Dp);
      // out_proj accumulated into tokens (residual + both directions)
      gemm(stream, ys, DI, ow, DI, ob, tokens, DM, M, DM, DI, ACT_NONE, 1);
    }

    // FFN block
    layernorm_kernel<<<M, 256, 0, stream>>>(tokens, fnorm_w + l * DM, fnorm_b + l * DM, yn);
    gemm(stream, yn, DM, ffn_w1 + (size_t)l * DFF * DM, DM, ffn_b1 + l * DFF,
         SCR, DFF, M, DFF, DM, ACT_GELU, 0);
    gemm(stream, SCR, DFF, ffn_w2 + (size_t)l * DM * DFF, DFF, ffn_b2 + l * DM,
         tokens, DM, M, DM, DFF, ACT_NONE, 1);
  }

  // final LN + head
  layernorm_kernel<<<M, 256, 0, stream>>>(tokens, norm_w, norm_b, yn);
  gemm(stream, yn, DM, head_w, DM, head_b, pred, Hq, M, Hq, DM, ACT_NONE, 0);
  head_out_kernel<<<(Bq * Hq * Nq + 255) / 256, 256, 0, stream>>>(pred, sd, mu, out);
}
